// GcnNet_47880295416003
// MI455X (gfx1250) — compile-verified, matrix operand order checked by reading, not run on hardware
//
#include <hip/hip_runtime.h>
#include <hip/hip_bf16.h>

// ---------------------------------------------------------------------------
// GCN 2-layer forward for MI455X (gfx1250, wave32, WMMA bf16).
//   support = X @ W1                (WMMA bf16, f32 accum)
//   h       = relu(A_sp @ support)  (warp-per-edge scatter, f32 atomics; relu
//                                    fused into GEMM2's A-tile conversion)
//   t       = relu(h) @ W2          (WMMA bf16, f32 accum)
//   logits  = A_sp @ t              (warp-per-edge scatter, f32 atomics)
// ---------------------------------------------------------------------------

typedef __attribute__((ext_vector_type(16))) __bf16        v16bf;
typedef __attribute__((ext_vector_type(8)))  float         v8f;
typedef __attribute__((ext_vector_type(8)))  unsigned int  v8u;

#define HID_DIM 128
#define OUT_DIM 64
#define IN_DIM  512

// round-to-nearest-even f32 -> bf16, packed pair (f0 in [15:0], f1 in [31:16])
__device__ __forceinline__ unsigned pack_bf16(float a, float b) {
    unsigned ua = __float_as_uint(a), ub = __float_as_uint(b);
    ua = (ua + 0x7FFFu + ((ua >> 16) & 1u)) >> 16;
    ub = (ub + 0x7FFFu + ((ub >> 16) & 1u)) >> 16;
    return (ub << 16) | (ua & 0xFFFFu);
}

// ---------------------------------------------------------------------------
// Tiled GEMM: C[nRows, NT*16] = op(A[nRows, K]) @ B[K, NT*16]
//   op = relu if RELU. BM = 128 rows per block, 256 threads = 8 waves.
//   Wave w computes rows [w*16, w*16+16) x all NT column tiles.
//   K stepped by 32 (one v_wmma_f32_16x16x32_bf16 per tile per step).
// LDS tiles hold packed bf16 pairs with stride 17 (bank-conflict-free:
// 17 coprime with 64 banks).
// ---------------------------------------------------------------------------
template <int NT, bool RELU>
__global__ __launch_bounds__(256) void gemm_bf16_wmma(
    const float* __restrict__ A, const float* __restrict__ B,
    float* __restrict__ C, int nRows, int K)
{
    constexpr int NB = NT * 16;                 // output columns
    __shared__ unsigned lds_a[128 * 17];        // 128 rows x 16 kpairs (+pad)
    __shared__ unsigned lds_b[NB * 17];         // NB  cols x 16 kpairs (+pad)

    const int tid   = threadIdx.x;
    const int wave  = tid >> 5;
    const int lane  = tid & 31;
    const int m     = lane & 15;                // M (or N) index inside tile
    const int khalf = lane >> 4;                // lane-half selects K group
    const int mBase = blockIdx.x * 128;

    v8f acc[NT];
#pragma unroll
    for (int c = 0; c < NT; ++c) acc[c] = v8f{};

    for (int kb = 0; kb < K; kb += 32) {
        __syncthreads();   // previous iteration's fragment loads done

        // ---- stage A tile: 128 rows x 32 k  (2048 packed pairs, 8/thread)
#pragma unroll
        for (int j = 0; j < 8; ++j) {
            int idx = tid + j * 256;            // 0..2047
            int row = idx >> 4;
            int kp  = idx & 15;
            int rg  = mBase + row;
            float f0 = 0.f, f1 = 0.f;
            if (rg < nRows) {
                const float2 f2 =
                    *(const float2*)(A + (size_t)rg * K + kb + kp * 2);
                f0 = f2.x; f1 = f2.y;
            }
            if (RELU) { f0 = fmaxf(f0, 0.f); f1 = fmaxf(f1, 0.f); }
            lds_a[row * 17 + kp] = pack_bf16(f0, f1);
        }

        // ---- stage B tile: 32 k x NB cols, packed per-column (NT/thread)
#pragma unroll
        for (int j = 0; j < NT; ++j) {
            int idx = tid + j * 256;            // 0..NB*16-1
            int col = idx % NB;
            int kp  = idx / NB;
            float f0 = B[(size_t)(kb + 2 * kp)     * NB + col];
            float f1 = B[(size_t)(kb + 2 * kp + 1) * NB + col];
            lds_b[col * 17 + kp] = pack_bf16(f0, f1);
        }

        __syncthreads();   // tiles visible

        // ---- A fragment for this wave's 16-row strip
        v8u au;
#pragma unroll
        for (int v = 0; v < 8; ++v) {
            int kp = (v & 3) + 4 * khalf + 8 * (v >> 2);   // ISA 16-bit A map
            au[v] = lds_a[(wave * 16 + m) * 17 + kp];
        }
        v16bf afrag = __builtin_bit_cast(v16bf, au);

        // ---- NT column tiles: load B frag, one WMMA each
#pragma unroll
        for (int c = 0; c < NT; ++c) {
            v8u bu;
#pragma unroll
            for (int v = 0; v < 8; ++v) {
                int kp = (v & 3) + 4 * khalf + 8 * (v >> 2);
                bu[v] = lds_b[(c * 16 + m) * 17 + kp];
            }
            v16bf bfrag = __builtin_bit_cast(v16bf, bu);
            acc[c] = __builtin_amdgcn_wmma_f32_16x16x32_bf16(
                false, afrag, false, bfrag, (short)0, acc[c], false, false);
        }
    }

    // ---- store D: VGPR r -> M = r + 8*khalf, N = lane%16
#pragma unroll
    for (int r = 0; r < 8; ++r) {
        int rg = mBase + wave * 16 + r + 8 * khalf;
        if (rg < nRows) {
            float* crow = C + (size_t)rg * NB + m;
#pragma unroll
            for (int c = 0; c < NT; ++c) crow[c * 16] = acc[c][r];
        }
    }
}

// ---------------------------------------------------------------------------
// SpMM scatter: out[row[e], :] += val[e] * dense[col[e], :]
// One wave (32 lanes) per edge; each lane owns D/32 contiguous floats.
// Hardware f32 atomics (no return) -> global_atomic_add_f32.
// ---------------------------------------------------------------------------
template <int D>
__global__ __launch_bounds__(256) void spmm_scatter(
    const int* __restrict__ erow, const int* __restrict__ ecol,
    const float* __restrict__ eval, const float* __restrict__ dense,
    float* __restrict__ out, int nEdges)
{
    const int e = blockIdx.x * 8 + (threadIdx.x >> 5);
    if (e >= nEdges) return;
    const int lane = threadIdx.x & 31;
    const int r = erow[e];
    const int c = ecol[e];
    const float v = eval[e];

    if constexpr (D == 128) {
        const float4 d = *(const float4*)(dense + (size_t)c * 128 + lane * 4);
        float* o = out + (size_t)r * 128 + lane * 4;
        unsafeAtomicAdd(o + 0, v * d.x);
        unsafeAtomicAdd(o + 1, v * d.y);
        unsafeAtomicAdd(o + 2, v * d.z);
        unsafeAtomicAdd(o + 3, v * d.w);
    } else {  // D == 64
        const float2 d = *(const float2*)(dense + (size_t)c * 64 + lane * 2);
        float* o = out + (size_t)r * 64 + lane * 2;
        unsafeAtomicAdd(o + 0, v * d.x);
        unsafeAtomicAdd(o + 1, v * d.y);
    }
}

// ---------------------------------------------------------------------------
extern "C" void kernel_launch(void* const* d_in, const int* in_sizes, int n_in,
                              void* d_out, int out_size, void* d_ws, size_t ws_size,
                              hipStream_t stream) {
    const float* feature = (const float*)d_in[0];
    const int*   erow    = (const int*)  d_in[1];
    const int*   ecol    = (const int*)  d_in[2];
    const float* evals   = (const float*)d_in[3];
    const float* W1      = (const float*)d_in[4];
    const float* W2      = (const float*)d_in[5];
    float*       logits  = (float*)d_out;

    const int N = in_sizes[0] / IN_DIM;     // 100000 nodes
    const int E = in_sizes[1];              // 1.6M edges

    const size_t supportBytes = (size_t)N * HID_DIM * sizeof(float);
    char* ws      = (char*)d_ws;
    float* support = (float*)(ws);                       // [N,128]
    float* h       = (float*)(ws + supportBytes);        // [N,128]
    float* t       = (float*)(ws);                       // [N,64] aliases dead support

    const int gemmBlocks = (N + 127) / 128;
    const int edgeBlocks = (E + 7) / 8;

    // layer 1: support = X @ W1
    gemm_bf16_wmma<8, false><<<gemmBlocks, 256, 0, stream>>>(
        feature, W1, support, N, IN_DIM);

    // h = A_sp @ support   (relu deferred into GEMM2's A conversion)
    hipMemsetAsync(h, 0, supportBytes, stream);
    spmm_scatter<128><<<edgeBlocks, 256, 0, stream>>>(
        erow, ecol, evals, support, h, E);

    // layer 2: t = relu(h) @ W2
    gemm_bf16_wmma<4, true><<<gemmBlocks, 256, 0, stream>>>(
        h, W2, t, N, HID_DIM);

    // logits = A_sp @ t
    hipMemsetAsync(logits, 0, (size_t)N * OUT_DIM * sizeof(float), stream);
    spmm_scatter<64><<<edgeBlocks, 256, 0, stream>>>(
        erow, ecol, evals, t, logits, E);
}
